// xPosAttention_1443109011505
// MI455X (gfx1250) — compile-verified, matrix operand order checked by reading, not run on hardware
//
#include <hip/hip_runtime.h>

// ---------------------------------------------------------------------------
// xPos attention for MI455X (gfx1250), bf16 WMMA pipeline with async-to-LDS
// double-buffered K/V staging in the flash-attention kernel.
// ---------------------------------------------------------------------------

typedef __attribute__((ext_vector_type(16))) __bf16 v16bf;
typedef __attribute__((ext_vector_type(8)))  float  v8f;

constexpr int BB  = 2;     // batch
constexpr int SS  = 2048;  // sequence
constexpr int DD  = 1024;  // model dim
constexpr int HH  = 16;    // heads
constexpr int DHH = 64;    // head dim

union frag16 {
  v16bf v;
  unsigned u[8];
};

__device__ __forceinline__ unsigned short f2bf(float f) {
  unsigned u = __float_as_uint(f);
  u += 0x7FFFu + ((u >> 16) & 1u);   // round-to-nearest-even
  return (unsigned short)(u >> 16);
}

// K-pair base for VGPR i of a 16-bit A/B fragment (ISA 7.12.2):
// VGPR0..3: K = 2i   (+8 for lanes 16-31); VGPR4..7: K = 16+2(i-4) (+8).
__device__ __forceinline__ int kmap(int i, int hlf) {
  return (i < 4) ? (2 * i + 8 * hlf) : (16 + 2 * (i - 4) + 8 * hlf);
}

__device__ __forceinline__ v8f wmma_bf16(frag16 a, frag16 b, v8f c) {
  return __builtin_amdgcn_wmma_f32_16x16x32_bf16(false, a.v, false, b.v,
                                                 (short)0, c, false, false);
}

__device__ __forceinline__ float rmax16(float v) {
#pragma unroll
  for (int off = 1; off < 16; off <<= 1) v = fmaxf(v, __shfl_xor(v, off, 32));
  return v;
}
__device__ __forceinline__ float rsum16(float v) {
#pragma unroll
  for (int off = 1; off < 16; off <<= 1) v += __shfl_xor(v, off, 32);
  return v;
}

// --- CDNA5 async global -> LDS copy (ASYNCcnt path), with sync fallback ----
#if defined(__has_builtin)
#if __has_builtin(__builtin_amdgcn_global_load_async_to_lds_b128)
#define ASYNC_LDS 1
#endif
#endif
#ifndef ASYNC_LDS
#define ASYNC_LDS 0
#endif

#if ASYNC_LDS
// ROCm clang declares the builtin with v4i pointees:
//   (v4i addrspace(1)* src, v4i addrspace(3)* dst, imm offset, imm cpol)
typedef int v4i_ __attribute__((ext_vector_type(4)));
typedef __attribute__((address_space(1))) v4i_ gas_v4i;
typedef __attribute__((address_space(3))) v4i_ las_v4i;
#endif

__device__ __forceinline__ void cp16(const unsigned short* g,
                                     unsigned short* l) {
#if ASYNC_LDS
  __builtin_amdgcn_global_load_async_to_lds_b128((gas_v4i*)(g), (las_v4i*)(l),
                                                 0, 0);
#else
  *(uint4*)l = *(const uint4*)g;
#endif
}

#if ASYNC_LDS
#define WAIT_ASYNC_LE8() asm volatile("s_wait_asynccnt 0x8" ::: "memory")
#define WAIT_ASYNC_0()   asm volatile("s_wait_asynccnt 0x0" ::: "memory")
#else
#define WAIT_ASYNC_LE8() do {} while (0)
#define WAIT_ASYNC_0()   do {} while (0)
#endif

// ---------------------------------------------------------------------------
// Kernel 1: fused QKV projection + xPos rotation.
//   mode (blockIdx.z): 0 = Q (Wq,bq, scale)  1 = K (Wv,bv, inv-scale)
//                      2 = V (Wk,bk, plain)        [reference swaps k/v!]
// Block = 256 threads (8 waves), block tile 64(M) x 64(N), K-chunk 32.
// ---------------------------------------------------------------------------
__global__ __launch_bounds__(256) void proj_qkv_kernel(
    const float* __restrict__ x,
    const float* __restrict__ Wq, const float* __restrict__ bq,
    const float* __restrict__ Wk, const float* __restrict__ bk,
    const float* __restrict__ Wv, const float* __restrict__ bv,
    unsigned short* __restrict__ Qb, unsigned short* __restrict__ Kb,
    unsigned short* __restrict__ Vb) {
  const int mode = blockIdx.z;
  const float* W; const float* bias; unsigned short* dst;
  if (mode == 0)      { W = Wq; bias = bq; dst = Qb; }
  else if (mode == 1) { W = Wv; bias = bv; dst = Kb; }
  else                { W = Wk; bias = bk; dst = Vb; }

  __shared__ __align__(16) unsigned short As[64 * 40];
  __shared__ __align__(16) unsigned short Bs[64 * 40];

  const int tid  = threadIdx.x;
  const int wave = tid >> 5, lane = tid & 31;
  const int wm = wave & 3, wn = wave >> 2;
  const int hlf = lane >> 4, l16 = lane & 15;
  const int m0 = blockIdx.y * 64;
  const int n0 = blockIdx.x * 64;
  const int lrow = tid >> 2;           // 0..63
  const int lcol = (tid & 3) * 8;      // 0,8,16,24

  v8f acc0 = {}; v8f acc1 = {};

  for (int k0 = 0; k0 < DD; k0 += 32) {
    __syncthreads();
    {  // stage A (x) tile as bf16
      const float* src = x + (size_t)(m0 + lrow) * DD + k0 + lcol;
      float4 f0 = *(const float4*)src;
      float4 f1 = *(const float4*)(src + 4);
      unsigned short* d = As + lrow * 40 + lcol;
      d[0] = f2bf(f0.x); d[1] = f2bf(f0.y); d[2] = f2bf(f0.z); d[3] = f2bf(f0.w);
      d[4] = f2bf(f1.x); d[5] = f2bf(f1.y); d[6] = f2bf(f1.z); d[7] = f2bf(f1.w);
    }
    {  // stage B (W row n) tile as bf16: B[n][k] = W[n][k]  (out = x @ W^T)
      const float* src = W + (size_t)(n0 + lrow) * DD + k0 + lcol;
      float4 f0 = *(const float4*)src;
      float4 f1 = *(const float4*)(src + 4);
      unsigned short* d = Bs + lrow * 40 + lcol;
      d[0] = f2bf(f0.x); d[1] = f2bf(f0.y); d[2] = f2bf(f0.z); d[3] = f2bf(f0.w);
      d[4] = f2bf(f1.x); d[5] = f2bf(f1.y); d[6] = f2bf(f1.z); d[7] = f2bf(f1.w);
    }
    __syncthreads();

    frag16 a, b0, b1;
#pragma unroll
    for (int i = 0; i < 8; ++i) {
      int k = kmap(i, hlf);
      a.u[i]  = *(const unsigned*)&As[(wm * 16 + l16) * 40 + k];
      b0.u[i] = *(const unsigned*)&Bs[(wn * 32 + l16) * 40 + k];
      b1.u[i] = *(const unsigned*)&Bs[(wn * 32 + 16 + l16) * 40 + k];
    }
    acc0 = wmma_bf16(a, b0, acc0);
    acc1 = wmma_bf16(a, b1, acc1);
  }

  // Epilogue: bias + (for Q/K) interleaved xPos rotation, store bf16 [B,H,S,DH].
  const float psign = (mode == 0) ? 1.f : -1.f;   // Q: scale, K: 1/scale
#pragma unroll
  for (int t = 0; t < 2; ++t) {
    v8f acc = t ? acc1 : acc0;
    const int n = n0 + wn * 32 + t * 16 + l16;
    const float bval = bias[n];
    const int h = n >> 6, d = n & 63;
#pragma unroll
    for (int j = 0; j < 8; ++j) {
      const int m    = m0 + wm * 16 + j + 8 * hlf;
      const int bidx = m >> 11;          // m / SS
      const int sidx = m & (SS - 1);
      float val = acc[j] + bval;
      unsigned short* outp =
          dst + ((((size_t)bidx * HH + h) * SS + sidx) * DHH + d);
      // rotate-half partner (d ^ 1) lives in neighbor lane (N = lane&15)
      float partner = __shfl_xor(val, 1, 32);
      if (mode == 2) {
        *outp = f2bf(val);
      } else {
        const int j2 = d >> 1;
        // inv_freq = theta^(-2*j2/64) = exp(-j2 * ln(1e4)/32)
        const float invf = __expf((float)j2 * -0.28782314f);
        const float ang  = (float)sidx * invf;
        const float c = cosf(ang), s = sinf(ang);
        // base = (2*(d%32) + 0.4*64) / (1.4*64); base in (0.28, 1)
        const float base  = (float)(d & 31) * 0.022321429f + 0.28571429f;
        const float power = ((float)sidx - 1024.0f) * (1.0f / 512.0f) * psign;
        const float scale = __expf(power * __logf(base));
        const float rh = (d & 1) ? partner : -partner;
        *outp = f2bf((val * c + rh * s) * scale);
      }
    }
  }
}

// ---------------------------------------------------------------------------
// Kernel 2: causal flash attention per (b, h, 64-row q-block).
// 128 threads = 4 waves; wave w owns q rows [qb*64 + w*16, +16).
// K/V tiles double-buffered via async global->LDS (ASYNCcnt pipeline).
// ---------------------------------------------------------------------------
__global__ __launch_bounds__(128) void attn_kernel(
    const unsigned short* __restrict__ Qb,
    const unsigned short* __restrict__ Kb,
    const unsigned short* __restrict__ Vb,
    unsigned short* __restrict__ interb) {
  const int qb = blockIdx.x, h = blockIdx.y, b = blockIdx.z;
  const int tid = threadIdx.x, wave = tid >> 5, lane = tid & 31;
  const int hlf = lane >> 4, l16 = lane & 15;

  __shared__ __align__(16) unsigned short Ks[2][64 * 72];
  __shared__ __align__(16) unsigned short Vs[2][64 * 72];
  __shared__ __align__(16) unsigned short Ps[64 * 72];

  const size_t headoff = ((size_t)b * HH + h) * SS * DHH;
  const unsigned short* Qh = Qb + headoff;
  const unsigned short* Kh = Kb + headoff;
  const unsigned short* Vh = Vb + headoff;

  // Stage one 64x64 bf16 K and V tile into buffer `buf`.
  // 8 async b128 instructions per thread-wave per tile (in-order completion).
  auto stage_tile = [&](int kb, int buf) {
#pragma unroll
    for (int i = 0; i < 4; ++i) {
      int idx = tid + i * 128;             // 0..511
      int row = idx >> 3, seg = (idx & 7) * 8;
      cp16(Kh + (size_t)(kb * 64 + row) * DHH + seg, &Ks[buf][row * 72 + seg]);
      cp16(Vh + (size_t)(kb * 64 + row) * DHH + seg, &Vs[buf][row * 72 + seg]);
    }
  };

  // Preload Q A-fragments (two K-chunks covering DH=64).
  frag16 qf[2];
  const int qrow = qb * 64 + wave * 16 + l16;
#pragma unroll
  for (int c = 0; c < 2; ++c)
#pragma unroll
    for (int i = 0; i < 8; ++i) {
      int k = c * 32 + kmap(i, hlf);
      qf[c].u[i] = *(const unsigned*)(Qh + (size_t)qrow * DHH + k);
    }

  v8f o[4] = {{}, {}, {}, {}};
  float mstat[8], lstat[8];
#pragma unroll
  for (int j = 0; j < 8; ++j) { mstat[j] = -3.0e38f; lstat[j] = 0.f; }

  stage_tile(0, 0);   // prologue of the async pipeline

  for (int kb = 0; kb <= qb; ++kb) {
    const int cur = kb & 1;
    __syncthreads();                 // buffer 1-cur no longer read by anyone
    if (kb + 1 <= qb) {
      stage_tile(kb + 1, 1 - cur);   // prefetch next tile while we compute
      WAIT_ASYNC_LE8();              // tile kb (first 8 asyncs) has landed
    } else {
      WAIT_ASYNC_0();
    }
    __syncthreads();                 // tile kb visible to all waves

    // S-tile: 16 q-rows x 64 keys = 4 WMMA tiles per wave.
    v8f st[4] = {{}, {}, {}, {}};
#pragma unroll
    for (int c = 0; c < 2; ++c) {
#pragma unroll
      for (int t = 0; t < 4; ++t) {
        frag16 bf;
        const int key = t * 16 + l16;        // B fragment: lane = N = key
#pragma unroll
        for (int i = 0; i < 8; ++i) {
          int k = c * 32 + kmap(i, hlf);
          bf.u[i] = *(const unsigned*)&Ks[cur][key * 72 + k];
        }
        st[t] = wmma_bf16(qf[c], bf, st[t]);
      }
    }

    // Scale 1/sqrt(DH*H)=1/32 and causal mask on the diagonal block.
#pragma unroll
    for (int t = 0; t < 4; ++t)
#pragma unroll
      for (int j = 0; j < 8; ++j) st[t][j] *= 0.03125f;
    if (kb == qb) {
#pragma unroll
      for (int t = 0; t < 4; ++t) {
        const int kg = kb * 64 + t * 16 + l16;
#pragma unroll
        for (int j = 0; j < 8; ++j) {
          const int qg = qb * 64 + wave * 16 + j + 8 * hlf;
          if (kg > qg) st[t][j] = -3.0e38f;
        }
      }
    }

    // Online softmax: row j+8*hlf stats reduced across the 16-lane half.
#pragma unroll
    for (int j = 0; j < 8; ++j) {
      float tmax = fmaxf(fmaxf(st[0][j], st[1][j]), fmaxf(st[2][j], st[3][j]));
      tmax = rmax16(tmax);
      const float mnew  = fmaxf(mstat[j], tmax);
      const float alpha = __expf(mstat[j] - mnew);
      float psum = 0.f;
#pragma unroll
      for (int t = 0; t < 4; ++t) {
        float p = __expf(st[t][j] - mnew);
        st[t][j] = p;
        psum += p;
      }
      psum = rsum16(psum);
      lstat[j] = lstat[j] * alpha + psum;
      mstat[j] = mnew;
#pragma unroll
      for (int t = 0; t < 4; ++t) o[t][j] *= alpha;
    }

    // Spill P (C-layout) to the wave-private LDS strip, bf16.
#pragma unroll
    for (int t = 0; t < 4; ++t)
#pragma unroll
      for (int j = 0; j < 8; ++j)
        Ps[(wave * 16 + j + 8 * hlf) * 72 + t * 16 + l16] = f2bf(st[t][j]);
    // Wave-private strip: this wave's DS writes -> its own reads only.
    asm volatile("s_wait_dscnt 0x0" ::: "memory");

    // O += P V  (A = P 16x32 per chunk; B = V keys x dh).
#pragma unroll
    for (int c = 0; c < 2; ++c) {
      frag16 pf;
#pragma unroll
      for (int i = 0; i < 8; ++i) {
        int k = c * 32 + kmap(i, hlf);
        pf.u[i] = *(const unsigned*)&Ps[(wave * 16 + l16) * 72 + k];
      }
#pragma unroll
      for (int t = 0; t < 4; ++t) {
        frag16 vf;
        const int dh = t * 16 + l16;
#pragma unroll
        for (int i = 0; i < 8; ++i) {
          int k = c * 32 + kmap(i, hlf);     // key pair (k, k+1), same dh col
          unsigned lo = Vs[cur][(unsigned)k * 72 + dh];
          unsigned hi = Vs[cur][(unsigned)(k + 1) * 72 + dh];
          vf.u[i] = lo | (hi << 16);
        }
        o[t] = wmma_bf16(pf, vf, o[t]);
      }
    }
  }

  // Normalize and scatter to inter[b, s, h*64 + dh] (bf16, merged-head layout).
#pragma unroll
  for (int j = 0; j < 8; ++j) {
    const float inv = 1.0f / lstat[j];
    const int qg = qb * 64 + wave * 16 + j + 8 * hlf;
    const size_t rowoff = ((size_t)b * SS + qg) * DD + h * DHH;
#pragma unroll
    for (int t = 0; t < 4; ++t)
      interb[rowoff + t * 16 + l16] = f2bf(o[t][j] * inv);
  }
}

// ---------------------------------------------------------------------------
// Kernel 3: output projection  out = inter @ Wo^T + bo  (bf16 A, f32 out).
// ---------------------------------------------------------------------------
__global__ __launch_bounds__(256) void oproj_kernel(
    const unsigned short* __restrict__ interb,
    const float* __restrict__ Wo, const float* __restrict__ bo,
    float* __restrict__ out) {
  __shared__ __align__(16) unsigned short As[64 * 40];
  __shared__ __align__(16) unsigned short Bs[64 * 40];

  const int tid  = threadIdx.x;
  const int wave = tid >> 5, lane = tid & 31;
  const int wm = wave & 3, wn = wave >> 2;
  const int hlf = lane >> 4, l16 = lane & 15;
  const int m0 = blockIdx.y * 64;
  const int n0 = blockIdx.x * 64;
  const int lrow = tid >> 2;
  const int lcol = (tid & 3) * 8;

  v8f acc0 = {}; v8f acc1 = {};

  for (int k0 = 0; k0 < DD; k0 += 32) {
    __syncthreads();
    {  // A already bf16: 4 uint4 per 32-element row chunk
      const int seg = tid & 3;
      const uint4* src =
          (const uint4*)(interb + (size_t)(m0 + (tid >> 2)) * DD + k0) + seg;
      *((uint4*)(As + (tid >> 2) * 40) + seg) = *src;
    }
    {  // B: Wo fp32 -> bf16
      const float* src = Wo + (size_t)(n0 + lrow) * DD + k0 + lcol;
      float4 f0 = *(const float4*)src;
      float4 f1 = *(const float4*)(src + 4);
      unsigned short* d = Bs + lrow * 40 + lcol;
      d[0] = f2bf(f0.x); d[1] = f2bf(f0.y); d[2] = f2bf(f0.z); d[3] = f2bf(f0.w);
      d[4] = f2bf(f1.x); d[5] = f2bf(f1.y); d[6] = f2bf(f1.z); d[7] = f2bf(f1.w);
    }
    __syncthreads();

    frag16 a, b0, b1;
#pragma unroll
    for (int i = 0; i < 8; ++i) {
      int k = kmap(i, hlf);
      a.u[i]  = *(const unsigned*)&As[(wm * 16 + l16) * 40 + k];
      b0.u[i] = *(const unsigned*)&Bs[(wn * 32 + l16) * 40 + k];
      b1.u[i] = *(const unsigned*)&Bs[(wn * 32 + 16 + l16) * 40 + k];
    }
    acc0 = wmma_bf16(a, b0, acc0);
    acc1 = wmma_bf16(a, b1, acc1);
  }

#pragma unroll
  for (int t = 0; t < 2; ++t) {
    v8f acc = t ? acc1 : acc0;
    const int n = n0 + wn * 32 + t * 16 + l16;
    const float bval = bo[n];
#pragma unroll
    for (int j = 0; j < 8; ++j) {
      const int m = m0 + wm * 16 + j + 8 * hlf;
      out[(size_t)m * DD + n] = acc[j] + bval;
    }
  }
}

// ---------------------------------------------------------------------------
extern "C" void kernel_launch(void* const* d_in, const int* in_sizes, int n_in,
                              void* d_out, int out_size, void* d_ws,
                              size_t ws_size, hipStream_t stream) {
  (void)in_sizes; (void)n_in; (void)out_size; (void)ws_size;
  const float* x  = (const float*)d_in[0];
  const float* Wq = (const float*)d_in[1];
  const float* bq = (const float*)d_in[2];
  const float* Wk = (const float*)d_in[3];
  const float* bk = (const float*)d_in[4];
  const float* Wv = (const float*)d_in[5];
  const float* bv = (const float*)d_in[6];
  const float* Wo = (const float*)d_in[7];
  const float* bo = (const float*)d_in[8];
  float* out = (float*)d_out;

  const size_t headElems = (size_t)BB * HH * SS * DHH;   // 4,194,304
  unsigned short* Qb     = (unsigned short*)d_ws;
  unsigned short* Kb     = Qb + headElems;
  unsigned short* Vb     = Kb + headElems;
  unsigned short* interb = Vb + headElems;               // BB*SS*DD bf16

  dim3 g1(DD / 64, (BB * SS) / 64, 3);
  proj_qkv_kernel<<<g1, 256, 0, stream>>>(x, Wq, bq, Wk, bk, Wv, bv, Qb, Kb, Vb);

  dim3 g2(SS / 64, HH, BB);
  attn_kernel<<<g2, 128, 0, stream>>>(Qb, Kb, Vb, interb);

  dim3 g3(DD / 64, (BB * SS) / 64, 1);
  oproj_kernel<<<g3, 256, 0, stream>>>(interb, Wo, bo, out);
}
